// SCoReGNN_67396626809054
// MI455X (gfx1250) — matrix operand order repaired and checked by reading, not compiled, not run on hardware
//
#include <hip/hip_runtime.h>
#include <math.h>
#include <stddef.h>

// ---------------------------------------------------------------------------
// Types for CDNA5 WMMA
// ---------------------------------------------------------------------------
typedef __attribute__((ext_vector_type(16))) _Float16 v16h;
typedef __attribute__((ext_vector_type(8)))  float    v8f;

#define TILE_M 128
#define TILE_N 64
#define TILE_K 32

// CDNA5 async global->LDS staging (ASYNCcnt path), with compile-safe fallback
#if defined(__gfx1250__) && __has_builtin(__builtin_amdgcn_global_load_async_to_lds_b128)
#define USE_ASYNC_LDS 1
#else
#define USE_ASYNC_LDS 0
#endif

#if USE_ASYNC_LDS
// Matches the builtin's parameter types: generic int4* source, LDS int4* dest.
typedef int v4i_ls __attribute__((__vector_size__(4 * sizeof(int))));
typedef __attribute__((address_space(3))) v4i_ls* lds_v4i_ptr;
__device__ __forceinline__ void async_copy_b128(const void* src, void* lds_dst) {
  __builtin_amdgcn_global_load_async_to_lds_b128(
      (v4i_ls*)src, (lds_v4i_ptr)lds_dst, 0, 0);
}
__device__ __forceinline__ void async_wait0() {
#if __has_builtin(__builtin_amdgcn_s_wait_asynccnt)
  __builtin_amdgcn_s_wait_asynccnt(0);
#else
  asm volatile("s_wait_asynccnt 0x0" ::: "memory");
#endif
}
#endif

// ---------------------------------------------------------------------------
// Utility kernels
// ---------------------------------------------------------------------------
__global__ void zero_kernel(float* p, long long n) {
  long long i = (long long)blockIdx.x * blockDim.x + threadIdx.x;
  if (i < n) p[i] = 0.0f;
}

__global__ void f32_to_f16_kernel(const float* __restrict__ X, _Float16* __restrict__ Y,
                                  long long n) {
  long long i = (long long)blockIdx.x * blockDim.x + threadIdx.x;
  long long stride = (long long)gridDim.x * blockDim.x;
  for (; i < n; i += stride) Y[i] = (_Float16)X[i];
}

__global__ void deg_kernel(const int* __restrict__ row, float* __restrict__ deg, int E) {
  int e = blockIdx.x * blockDim.x + threadIdx.x;
  if (e < E) atomicAdd(&deg[row[e]], 1.0f);
}

// ---------------------------------------------------------------------------
// WMMA GEMM:  Y[n, j] = sum_c A[n, c] * W[j, c] + bias[j]   (+ optional ReLU)
// A: nrows x kdim (row major, f16), W: dout x kdim (row major, f16)
// Block: 256 threads (8 waves). Block tile 128x64; wave tile 16x64.
// Staging: branchless 16B chunks; async global->LDS when available.
// Requires: kdim % 32 == 0, dout % 64 == 0 (true for all call sites).
// ---------------------------------------------------------------------------
__global__ __launch_bounds__(256) void gemm_wmma_kernel(
    const _Float16* __restrict__ A, const _Float16* __restrict__ Wt,
    const float* __restrict__ bias, float* __restrict__ Y,
    int nrows, int kdim, int dout, int do_relu) {
  __shared__ _Float16 sA[TILE_M][TILE_K + 8];   // row stride 80B (16B aligned)
  __shared__ _Float16 sB[TILE_N][TILE_K + 8];

  const int tid  = threadIdx.x;
  const int lane = tid & 31;
  const int wave = tid >> 5;                 // 0..7
  const int rowBase = blockIdx.x * TILE_M;
  const int colBase = blockIdx.y * TILE_N;

  v8f acc[4] = {};

  for (int k0 = 0; k0 < kdim; k0 += TILE_K) {
    // ---- Stage A tile: 128 rows x 32 halves = 512 chunks of 16B ----
#pragma unroll
    for (int it = 0; it < 2; ++it) {
      int q = tid + it * 256;
      int r = q >> 2;
      int c = (q & 3) * 8;
      int gr = min(rowBase + r, nrows - 1);   // clamp: only affects rows never stored
      const _Float16* src = &A[(size_t)gr * kdim + k0 + c];
      _Float16* dst = &sA[r][c];
#if USE_ASYNC_LDS
      async_copy_b128(src, dst);
#else
      *(uint4*)dst = *(const uint4*)src;
#endif
    }
    // ---- Stage B tile: 64 rows x 32 halves = 256 chunks of 16B ----
    {
      int r = tid >> 2;
      int c = (tid & 3) * 8;
      const _Float16* src = &Wt[(size_t)(colBase + r) * kdim + k0 + c];
      _Float16* dst = &sB[r][c];
#if USE_ASYNC_LDS
      async_copy_b128(src, dst);
#else
      *(uint4*)dst = *(const uint4*)src;
#endif
    }
#if USE_ASYNC_LDS
    async_wait0();
#else
    // prefetch next A tile into cache hierarchy (global_prefetch_b8)
    if (k0 + TILE_K < kdim) {
      int r = min(rowBase + (tid >> 1), nrows - 1);
      __builtin_prefetch(&A[(size_t)r * kdim + k0 + TILE_K + (tid & 1) * 16], 0, 1);
    }
#endif
    __syncthreads();

    // Build A fragment: 16x32 f16, M = lane&15,
    // K split: lanes<16 hold K 0..15; lanes>=16 hold K 16..31 (matches B)
    const int mrow  = wave * 16 + (lane & 15);
    const int koffA = (lane & 16) ? 16 : 0;
    v16h a;
#pragma unroll
    for (int e = 0; e < 16; ++e) {
      int kk = e + koffA;
      a[e] = sA[mrow][kk];
    }

    // Build 4 B fragments (cols j*16..j*16+15): lanes<16 -> K 0..15, lanes>=16 -> K 16..31
    const int bcol  = lane & 15;
    const int koffB = (lane & 16) ? 16 : 0;
#pragma unroll
    for (int j = 0; j < 4; ++j) {
      v16h b;
#pragma unroll
      for (int e = 0; e < 16; ++e) b[e] = sB[j * 16 + bcol][koffB + e];
      acc[j] = __builtin_amdgcn_wmma_f32_16x16x32_f16(
          false, a, false, b, (short)0, acc[j], false, false);
    }
    __syncthreads();
  }

  // Store D: VGPR r -> M = r + (lane>=16 ? 8 : 0); N = lane&15
  const int mrow0 = rowBase + wave * 16 + ((lane & 16) ? 8 : 0);
  const int ncol0 = colBase + (lane & 15);
#pragma unroll
  for (int j = 0; j < 4; ++j) {
    int col = ncol0 + j * 16;
    float bv = bias[col];
#pragma unroll
    for (int r = 0; r < 8; ++r) {
      int gr = mrow0 + r;
      if (gr < nrows) {
        float v = acc[j][r] + bv;
        if (do_relu) v = fmaxf(v, 0.0f);
        Y[(size_t)gr * dout + col] = v;
      }
    }
  }
}

// ---------------------------------------------------------------------------
// Probe RNG (counter-based hash + Box-Muller) and trace estimation.
// trace = sum_n deg[n]*||P[n,:]||^2  -  sum_e P[row_e,:].P[col_e,:]
// (algebraic elimination of the ap segment-sum in the reference)
// ---------------------------------------------------------------------------
__device__ __forceinline__ unsigned hash_u32(unsigned x) {
  x ^= x >> 16; x *= 0x7feb352dU;
  x ^= x >> 15; x *= 0x846ca68bU;
  x ^= x >> 16; return x;
}

__device__ __forceinline__ float probe_val(unsigned seed, unsigned idx) {
  unsigned h1 = hash_u32(idx ^ (seed * 0x9E3779B9u) ^ 0xA5A5A5A5u);
  unsigned h2 = hash_u32(h1 + 0x85EBCA6Bu);
  float u1 = (float)(h1 >> 8) * (1.0f / 16777216.0f) + (0.5f / 16777216.0f);
  float u2 = (float)(h2 >> 8) * (1.0f / 16777216.0f);
  return __fsqrt_rn(-2.0f * __logf(u1)) * __cosf(6.28318530718f * u2);
}

__global__ void probe_gen_kernel(float* __restrict__ P, int n, unsigned seed) {
  long long t = (long long)blockIdx.x * blockDim.x + threadIdx.x;
  long long total = (long long)n * 256;
  if (t < total) P[t] = probe_val(seed, (unsigned)t);
}

__global__ void trace_deg_kernel(const float* __restrict__ P, const float* __restrict__ deg,
                                 float* __restrict__ acc, int n) {
  __shared__ float red[256];
  int i = blockIdx.x * blockDim.x + threadIdx.x;
  float s = 0.0f;
  if (i < n) {
    const float4* p = (const float4*)(P + (size_t)i * 256);
    float ss = 0.0f;
    for (int j = 0; j < 64; ++j) {
      float4 v = p[j];
      ss += v.x * v.x + v.y * v.y + v.z * v.z + v.w * v.w;
    }
    s = deg[i] * ss;
  }
  red[threadIdx.x] = s;
  __syncthreads();
  for (int st = 128; st > 0; st >>= 1) {
    if (threadIdx.x < st) red[threadIdx.x] += red[threadIdx.x + st];
    __syncthreads();
  }
  if (threadIdx.x == 0) atomicAdd(acc, red[0]);
}

__global__ void trace_edge_kernel(const float* __restrict__ P, const int* __restrict__ row,
                                  const int* __restrict__ col, float* __restrict__ acc, int E) {
  __shared__ float red[256];
  int e = blockIdx.x * blockDim.x + threadIdx.x;
  float s = 0.0f;
  if (e < E) {
    const float4* pr = (const float4*)(P + (size_t)row[e] * 256);
    const float4* pc = (const float4*)(P + (size_t)col[e] * 256);
    float dot = 0.0f;
    for (int j = 0; j < 64; ++j) {
      float4 a = pr[j], b = pc[j];
      dot += a.x * b.x + a.y * b.y + a.z * b.z + a.w * b.w;
    }
    s = -dot;
  }
  red[threadIdx.x] = s;
  __syncthreads();
  for (int st = 128; st > 0; st >>= 1) {
    if (threadIdx.x < st) red[threadIdx.x] += red[threadIdx.x + st];
    __syncthreads();
  }
  if (threadIdx.x == 0) atomicAdd(acc, red[0]);
}

__global__ void sigma_kernel(const float* __restrict__ acc, float* __restrict__ sigma) {
  if (threadIdx.x == 0 && blockIdx.x == 0) {
    float tr = acc[0] * (1.0f / 256.0f);
    float a = fmaxf(fabsf(tr), 1e-6f);
    float s = sqrtf(64.0f / a);
    sigma[0] = fminf(fmaxf(s, 0.1f), 10.0f);
  }
}

// ---------------------------------------------------------------------------
// Fastfood: Y = fwht(perm_gather(fwht(X*B)) * G) * S / sigma
// one node per block; blockDim == d; in-LDS butterflies
// ---------------------------------------------------------------------------
__global__ void fastfood_kernel(const float* __restrict__ X, float* __restrict__ Y,
                                const float* __restrict__ Bv, const float* __restrict__ G,
                                const float* __restrict__ S, const int* __restrict__ perm,
                                const float* __restrict__ sigma_p, int d, float inv_sqrt_d) {
  extern __shared__ float sm[];
  float* u = sm;
  float* t = sm + d;
  const int n = blockIdx.x;
  const int i = threadIdx.x;

  u[i] = X[(size_t)n * d + i] * Bv[i];
  __syncthreads();
  for (int h = 1; h < d; h <<= 1) {
    float a = u[i], b = u[i ^ h];
    float v = (i & h) ? (b - a) : (a + b);
    __syncthreads();
    u[i] = v;
    __syncthreads();
  }
  t[i] = u[perm[i]] * inv_sqrt_d * G[i];
  __syncthreads();
  for (int h = 1; h < d; h <<= 1) {
    float a = t[i], b = t[i ^ h];
    float v = (i & h) ? (b - a) : (a + b);
    __syncthreads();
    t[i] = v;
    __syncthreads();
  }
  Y[(size_t)n * d + i] = t[i] * inv_sqrt_d * S[i] / sigma_p[0];
}

// ---------------------------------------------------------------------------
// Attention: global-scalar softmax max, per-head denominators, edge scatter
// ---------------------------------------------------------------------------
__device__ __forceinline__ unsigned float_to_ord(float f) {
  unsigned u = __float_as_uint(f);
  return (u & 0x80000000u) ? ~u : (u | 0x80000000u);
}
__device__ __forceinline__ float ord_to_float(unsigned u) {
  u = (u & 0x80000000u) ? (u & 0x7fffffffu) : ~u;
  return __uint_as_float(u);
}

__global__ void alpha_kernel(const float* __restrict__ QP, const float* __restrict__ KP,
                             const int* __restrict__ row, const int* __restrict__ col,
                             float* __restrict__ alpha, int E, int N, int heads, int d,
                             unsigned* __restrict__ amax_ord) {
  __shared__ unsigned red[256];
  const int M = E + N;
  int e = blockIdx.x * blockDim.x + threadIdx.x;
  float lmax = -3.4e38f;
  if (e < M) {
    int src = (e < E) ? row[e] : (e - E);
    int dst = (e < E) ? col[e] : (e - E);
    for (int h = 0; h < heads; ++h) {
      const float4* qp = (const float4*)(QP + (size_t)dst * d + h * 64);
      const float4* kp = (const float4*)(KP + (size_t)src * d + h * 64);
      float s = 0.0f;
      for (int j = 0; j < 16; ++j) {
        float4 a = qp[j], b = kp[j];
        s += a.x * b.x + a.y * b.y + a.z * b.z + a.w * b.w;
      }
      s *= 0.125f;                       // 1/sqrt(64)
      s = (s > 0.0f) ? s : 0.2f * s;     // leaky_relu
      alpha[(size_t)h * M + e] = s;
      lmax = fmaxf(lmax, s);
    }
  }
  red[threadIdx.x] = float_to_ord(lmax);
  __syncthreads();
  for (int st = 128; st > 0; st >>= 1) {
    if (threadIdx.x < st) red[threadIdx.x] = max(red[threadIdx.x], red[threadIdx.x + st]);
    __syncthreads();
  }
  if (threadIdx.x == 0) atomicMax(amax_ord, red[0]);
}

__global__ void expnorm_kernel(float* __restrict__ alpha, const unsigned* __restrict__ amax_ord,
                               float* __restrict__ denom, int M, int heads) {
  __shared__ float red[256];
  int e = blockIdx.x * blockDim.x + threadIdx.x;
  float amax = ord_to_float(*amax_ord);
  float ls[4] = {0.0f, 0.0f, 0.0f, 0.0f};
  if (e < M) {
    for (int h = 0; h < heads; ++h) {
      float w = __expf(alpha[(size_t)h * M + e] - amax);
      alpha[(size_t)h * M + e] = w;
      ls[h] = w;
    }
  }
  for (int h = 0; h < heads; ++h) {
    red[threadIdx.x] = ls[h];
    __syncthreads();
    for (int st = 128; st > 0; st >>= 1) {
      if (threadIdx.x < st) red[threadIdx.x] += red[threadIdx.x + st];
      __syncthreads();
    }
    if (threadIdx.x == 0) atomicAdd(&denom[h], red[0]);
    __syncthreads();
  }
}

// One wave per (edge, head); 32 lanes cover 64 channels (2 each)
__global__ void scatter_kernel(const float* __restrict__ alpha, const float* __restrict__ denom,
                               const float* __restrict__ V, const int* __restrict__ row,
                               const int* __restrict__ col, float* __restrict__ H,
                               int E, int N, int heads, int d) {
  const int M = E + N;
  long long wid = ((long long)blockIdx.x * blockDim.x + threadIdx.x) >> 5;
  const int lane = threadIdx.x & 31;
  long long total = (long long)M * heads;
  if (wid >= total) return;
  int e = (int)(wid / heads);
  int h = (int)(wid % heads);
  int src = (e < E) ? row[e] : (e - E);
  int dst = (e < E) ? col[e] : (e - E);
  float coef = alpha[(size_t)h * M + e] / denom[h];
  size_t vb = (size_t)src * d + h * 64;
  size_t hb = (size_t)dst * d + h * 64;
  atomicAdd(&H[hb + lane],       V[vb + lane] * coef);
  atomicAdd(&H[hb + lane + 32],  V[vb + lane + 32] * coef);
}

// ---------------------------------------------------------------------------
// Host orchestration
// ---------------------------------------------------------------------------
static void run_scrf(const float* Qin, float* Qout, const float* Bv, const float* G,
                     const float* S, const int* perm, const int* row, const int* col,
                     const float* DEG, float* P, float* SC, int N, int E, int d,
                     unsigned seed, hipStream_t stream) {
  long long tot = (long long)N * 256;
  probe_gen_kernel<<<(unsigned)((tot + 255) / 256), 256, 0, stream>>>(P, N, seed);
  zero_kernel<<<1, 256, 0, stream>>>(SC, 1);
  trace_deg_kernel<<<(N + 255) / 256, 256, 0, stream>>>(P, DEG, SC, N);
  trace_edge_kernel<<<(E + 255) / 256, 256, 0, stream>>>(P, row, col, SC, E);
  sigma_kernel<<<1, 32, 0, stream>>>(SC, SC + 1);
  float isd = (float)(1.0 / sqrt((double)d));
  fastfood_kernel<<<N, d, 2 * d * sizeof(float), stream>>>(Qin, Qout, Bv, G, S, perm,
                                                           SC + 1, d, isd);
}

extern "C" void kernel_launch(void* const* d_in, const int* in_sizes, int n_in,
                              void* d_out, int out_size, void* d_ws, size_t ws_size,
                              hipStream_t stream) {
  (void)n_in; (void)out_size; (void)ws_size;

  const float* x_in = (const float*)d_in[0];
  const int*   edge = (const int*)d_in[1];
  const int N = in_sizes[0] / 256;
  const int E = in_sizes[1] / 2;
  const int* row = edge;
  const int* col = edge + E;

  // Workspace carve-up (floats, then f16 buffers at the tail)
  float* ws = (float*)d_ws;
  const size_t NS = (size_t)N * 256;
  float* Qb    = ws;                 // N x 256
  float* Kb    = Qb + NS;            // N x 256
  float* Vb    = Kb + NS;            // N x 256
  float* QP    = Vb + NS;            // N x 256
  float* KP    = QP + NS;            // N x 256
  float* Hb    = KP + NS;            // N x 256
  float* X1    = Hb + NS;            // N x 64
  float* X2    = X1 + (size_t)N * 64;
  float* P     = X2 + (size_t)N * 64; // N x 256 probe
  float* ALPHA = P + NS;             // heads x (E+N)
  float* DEG   = ALPHA + (size_t)(E + N) * 4;
  float* SC    = DEG + N;            // [0]=trace acc [1]=sigma [2..5]=denom [6]=amax ord
  _Float16* XA16 = (_Float16*)(SC + 64);      // N x 256 halves (A operand, reused for H)
  _Float16* W16q = XA16 + NS;                 // up to 256x256 halves each
  _Float16* W16k = W16q + 65536;
  _Float16* W16v = W16k + 65536;
  _Float16* Wo16 = W16v + 65536;

  // degrees (once, edge_index is layer-invariant)
  zero_kernel<<<(unsigned)((N + 255) / 256), 256, 0, stream>>>(DEG, N);
  deg_kernel<<<(E + 255) / 256, 256, 0, stream>>>(row, DEG, E);

  const int cin_tab[3]   = {256, 64, 64};
  const int heads_tab[3] = {4, 4, 1};

  const float* Xcur = x_in;
  for (int l = 0; l < 3; ++l) {
    const int cin   = cin_tab[l];
    const int heads = heads_tab[l];
    const int d     = heads * 64;
    const int base  = 2 + l * 11;

    const float* Wq = (const float*)d_in[base + 0];
    const float* bq = (const float*)d_in[base + 1];
    const float* Wk = (const float*)d_in[base + 2];
    const float* bk = (const float*)d_in[base + 3];
    const float* Wv = (const float*)d_in[base + 4];
    const float* bv = (const float*)d_in[base + 5];
    const float* Wo = (const float*)d_in[base + 6];
    const float* bo = (const float*)d_in[base + 7];
    const float* Bv = (const float*)d_in[base + 8];
    const float* G  = (const float*)d_in[base + 9];
    const float* S  = (const float*)d_in[base + 10];
    const int* perm = (const int*)d_in[35 + l];

    // f32 -> f16 conversions (X once for all of q/k/v; weights once each)
    f32_to_f16_kernel<<<2048, 256, 0, stream>>>(Xcur, XA16, (long long)N * cin);
    f32_to_f16_kernel<<<64, 256, 0, stream>>>(Wq, W16q, (long long)d * cin);
    f32_to_f16_kernel<<<64, 256, 0, stream>>>(Wk, W16k, (long long)d * cin);
    f32_to_f16_kernel<<<64, 256, 0, stream>>>(Wv, W16v, (long long)d * cin);
    f32_to_f16_kernel<<<64, 256, 0, stream>>>(Wo, Wo16, (long long)64 * d);

    // q/k/v projections (WMMA GEMMs)
    dim3 gg((N + TILE_M - 1) / TILE_M, d / TILE_N);
    gemm_wmma_kernel<<<gg, 256, 0, stream>>>(XA16, W16q, bq, Qb, N, cin, d, 0);
    gemm_wmma_kernel<<<gg, 256, 0, stream>>>(XA16, W16k, bk, Kb, N, cin, d, 0);
    gemm_wmma_kernel<<<gg, 256, 0, stream>>>(XA16, W16v, bv, Vb, N, cin, d, 0);

    // scrf random features for q and k (distinct probe streams)
    run_scrf(Qb, QP, Bv, G, S, perm, row, col, DEG, P, SC, N, E, d,
             0x1000u + (unsigned)l * 2u + 0u, stream);
    run_scrf(Kb, KP, Bv, G, S, perm, row, col, DEG, P, SC, N, E, d,
             0x1000u + (unsigned)l * 2u + 1u, stream);

    // attention
    const int M = E + N;
    zero_kernel<<<(unsigned)(((long long)N * d + 255) / 256), 256, 0, stream>>>(Hb, (long long)N * d);
    zero_kernel<<<1, 256, 0, stream>>>(SC + 2, 5); // denom[4] + amax ord (0.0f bits == 0u)
    alpha_kernel<<<(M + 255) / 256, 256, 0, stream>>>(QP, KP, row, col, ALPHA, E, N, heads, d,
                                                      (unsigned*)(SC + 6));
    expnorm_kernel<<<(M + 255) / 256, 256, 0, stream>>>(ALPHA, (const unsigned*)(SC + 6),
                                                        SC + 2, M, heads);
    long long sthreads = (long long)M * heads * 32;
    scatter_kernel<<<(unsigned)((sthreads + 255) / 256), 256, 0, stream>>>(
        ALPHA, SC + 2, Vb, row, col, Hb, E, N, heads, d);

    // output projection (+ReLU for layers 0,1); final layer writes d_out
    f32_to_f16_kernel<<<2048, 256, 0, stream>>>(Hb, XA16, (long long)N * d);
    float* outp = (l == 2) ? (float*)d_out : ((l == 0) ? X1 : X2);
    dim3 go((N + TILE_M - 1) / TILE_M, 1);
    gemm_wmma_kernel<<<go, 256, 0, stream>>>(XA16, Wo16, bo, outp, N, d, 64, (l < 2) ? 1 : 0);
    Xcur = outp;
  }
}
